// QPrior_25237227832011
// MI455X (gfx1250) — matrix operand order repaired and checked by reading, not compiled, not run on hardware
//
#include <hip/hip_runtime.h>
#include <hip/hip_bf16.h>
#include <math.h>

typedef __attribute__((ext_vector_type(16))) __bf16 v16bf;
typedef __attribute__((ext_vector_type(8)))  float  v8f;

#define E_  10
#define B_  4096
#define DS_ 256
#define DA_ 64
#define D_  320
#define H_  1024
#define K_  2
#define EB  (E_*B_)

// LDS row pitch (bf16 elems) for staged tiles: 64 data + 8 pad.
// 72*2 = 144 B pitch -> every 16B-chunk dest stays 16B aligned for async b128.
#define LPITCH 72

typedef __attribute__((address_space(3))) __bf16 lds_bf16_t;

// Async global->LDS 128-bit copy (CDNA5, tracked by ASYNCcnt).
__device__ __forceinline__ void async_b128(__bf16* lds_dst, const __bf16* gsrc) {
  lds_bf16_t* d = (lds_bf16_t*)lds_dst;   // addrspacecast: generic -> LDS (32-bit)
  asm volatile("global_load_async_to_lds_b128 %0, %1, off"
               :: "v"(d), "v"(gsrc) : "memory");
}
__device__ __forceinline__ void wait_async0() {
  asm volatile("s_wait_asynccnt 0x0" ::: "memory");
}

// ---------------------------------------------------------------------------
// 1) Build xu = concat(state, actions[e]) as bf16  [E, B, D]
// ---------------------------------------------------------------------------
__global__ __launch_bounds__(256)
void build_xu(const float* __restrict__ state, const float* __restrict__ actions,
              __bf16* __restrict__ xu) {
  int idx = blockIdx.x * blockDim.x + threadIdx.x;
  const int total = E_ * B_ * D_;
  const int stride = gridDim.x * blockDim.x;
  for (; idx < total; idx += stride) {
    int d  = idx % D_;
    int eb = idx / D_;
    int b  = eb % B_;
    int e  = eb / B_;
    float v = (d < DS_) ? state[b * DS_ + d]
                        : actions[(e * B_ + b) * DA_ + (d - DS_)];
    xu[idx] = (__bf16)v;
  }
}

// ---------------------------------------------------------------------------
// 2) Weight pre-pass: W fp32 [Kd x H_] (slice z) -> Wt bf16 [H_ x Kd] (transposed)
//    grid = (H_/64, Kd/64, nSlices); 64x64 tile through LDS, coalesced both ways.
// ---------------------------------------------------------------------------
__global__ __launch_bounds__(256)
void convert_transpose(const float* __restrict__ W, __bf16* __restrict__ Wt, int Kd) {
  __shared__ __bf16 t[64][65];
  const size_t zo = (size_t)blockIdx.z;
  const float* Ws = W  + zo * (size_t)Kd * H_;
  __bf16*     Wts = Wt + zo * (size_t)H_ * Kd;
  const int n0 = blockIdx.x * 64;
  const int k0 = blockIdx.y * 64;
  #pragma unroll
  for (int q = 0; q < 16; ++q) {
    int u = threadIdx.x + (q << 8);
    int r = u >> 6, c = u & 63;                 // r: k-row, c: n-col (coalesced read)
    t[c][r] = (__bf16)Ws[(size_t)(k0 + r) * H_ + n0 + c];
  }
  __syncthreads();
  #pragma unroll
  for (int q = 0; q < 16; ++q) {
    int u = threadIdx.x + (q << 8);
    int r = u >> 6, c = u & 63;                 // r: n-row, c: k-col (coalesced write)
    Wts[(size_t)(n0 + r) * Kd + k0 + c] = t[r][c];
  }
}

// ---------------------------------------------------------------------------
// 3) C = relu(A[M x Kd](bf16) * Wt[H_ x Kd](bf16,transposed)^T + bias) -> bf16
//    Block 256 (8 waves as 4(M) x 2(N)), tile 128x128, K-chunk 64,
//    double-buffered async LDS staging. Each wave: 2 M-subtiles x 4 N-subtiles
//    -> 6 fragment loads per 8 WMMAs (vs 9 for 1-D wave tiling).
//    grid = dim3(H_/128, B_/128)
// ---------------------------------------------------------------------------
__global__ __launch_bounds__(256)
void gemm_wmma_async(const __bf16* __restrict__ A, int lda, int Kd,
                     const __bf16* __restrict__ Wt,
                     const float* __restrict__ bias,
                     __bf16* __restrict__ Cout) {
  __shared__ __bf16 lds_a[2][128 * LPITCH];   // 18,432 B each
  __shared__ __bf16 lds_b[2][128 * LPITCH];   // total LDS: 73,728 B

  const int tid  = threadIdx.x;
  const int wave = tid >> 5;
  const int lane = tid & 31;
  const int hl   = lane >> 4;
  const int l16  = lane & 15;
  const int wm   = wave & 3;    // M group: rows 32*wm .. 32*wm+31
  const int wn   = wave >> 2;   // N group: cols 64*wn .. 64*wn+63
  const int tileN = blockIdx.x * 128;
  const int tileM = blockIdx.y * 128;

  v8f acc[2][4];
  #pragma unroll
  for (int i = 0; i < 2; ++i)
    #pragma unroll
    for (int n = 0; n < 4; ++n)
      #pragma unroll
      for (int j = 0; j < 8; ++j) acc[i][n][j] = 0.0f;

  // Issue async copies for one 128x64 A tile and one 128x64 Wt tile.
  auto stage = [&](int buf, int k0) {
    #pragma unroll
    for (int q = 0; q < 4; ++q) {
      int u   = tid + (q << 8);            // 0..1023 (16B chunks per matrix)
      int row = u >> 3;                    // 0..127
      int c   = (u & 7) * 8;               // bf16 col within chunk row
      async_b128(&lds_a[buf][row * LPITCH + c],
                 A  + (size_t)(tileM + row) * lda + k0 + c);
      async_b128(&lds_b[buf][row * LPITCH + c],
                 Wt + (size_t)(tileN + row) * Kd  + k0 + c);
    }
  };

  stage(0, 0);
  wait_async0();
  __syncthreads();

  const int nch = Kd >> 6;
  for (int kc = 0; kc < nch; ++kc) {
    const int buf = kc & 1;
    if (kc + 1 < nch) stage(buf ^ 1, (kc + 1) << 6);   // prefetch next chunk

    const __bf16* la = lds_a[buf];
    const __bf16* lb = lds_b[buf];
    #pragma unroll
    for (int s = 0; s < 2; ++s) {          // two K=32 sub-steps of the 64-chunk
      // A fragments for this wave's two 16-row subtiles (contiguous k order: k = 16*hl + element)
      union { unsigned int u[8]; v16bf v; } af[2];
      #pragma unroll
      for (int i = 0; i < 2; ++i) {
        int arow = wm * 32 + i * 16 + l16;
        #pragma unroll
        for (int j = 0; j < 8; ++j) {
          int kb = 32 * s + 16 * hl + 2 * j;
          af[i].u[j] = *(const unsigned int*)&la[arow * LPITCH + kb];
        }
      }
      // B fragments for this wave's four 16-col subtiles (contiguous k order: k = 16*hl + element)
      union { unsigned int u[8]; v16bf v; } bf[4];
      #pragma unroll
      for (int n = 0; n < 4; ++n) {
        int brow = wn * 64 + n * 16 + l16;
        #pragma unroll
        for (int j = 0; j < 8; ++j) {
          int kb = 32 * s + 16 * hl + 2 * j;
          bf[n].u[j] = *(const unsigned int*)&lb[brow * LPITCH + kb];
        }
      }
      #pragma unroll
      for (int i = 0; i < 2; ++i)
        #pragma unroll
        for (int n = 0; n < 4; ++n)
          acc[i][n] = __builtin_amdgcn_wmma_f32_16x16x32_bf16(
              false, af[i].v, false, bf[n].v, (short)0, acc[i][n], false, false);
    }
    wait_async0();       // next chunk's tiles landed
    __syncthreads();     // all waves done reading current buffers
  }

  // epilogue: bias + relu + bf16 store
  #pragma unroll
  for (int n = 0; n < 4; ++n) {
    int col = tileN + wn * 64 + n * 16 + l16;
    float bv = bias[col];
    #pragma unroll
    for (int i = 0; i < 2; ++i) {
      #pragma unroll
      for (int j = 0; j < 8; ++j) {
        int row = tileM + wm * 32 + i * 16 + hl * 8 + j;
        float v = acc[i][n][j] + bv;
        v = fmaxf(v, 0.0f);
        Cout[(size_t)row * H_ + col] = (__bf16)v;
      }
    }
  }
}

// ---------------------------------------------------------------------------
// 4) Layer-3 GEMV: res[r] = dot(h2[r,:], Wout) + bout   (one wave per row)
// ---------------------------------------------------------------------------
__global__ __launch_bounds__(256)
void head_out(const __bf16* __restrict__ h2, const float* __restrict__ Wout,
              const float* __restrict__ bout, float* __restrict__ res) {
  int wave = threadIdx.x >> 5;
  int lane = threadIdx.x & 31;
  int r = blockIdx.x * 8 + wave;
  float s = 0.0f;
  const __bf16* hrow = h2 + (size_t)r * H_;
  #pragma unroll 4
  for (int i = lane; i < H_; i += 32) s += (float)hrow[i] * Wout[i];
  #pragma unroll
  for (int off = 16; off > 0; off >>= 1) s += __shfl_xor(s, off, 32);
  if (lane == 0) res[r] = s + bout[0];
}

// ---------------------------------------------------------------------------
// 5) a1/a2/min + deterministic two-stage std reduction
// ---------------------------------------------------------------------------
__global__ __launch_bounds__(256)
void combine(const float* __restrict__ res, float* __restrict__ out,
             float* __restrict__ partials) {
  __shared__ float ss[256], sq[256];
  int tid = threadIdx.x;
  int i = blockIdx.x * 256 + tid;
  float a1 = res[i] + 3.0f * res[2 * EB + i];
  float a2 = res[EB + i] + 3.0f * res[3 * EB + i];
  float mv = fminf(a1, a2);
  out[i] = a1;
  out[EB + i] = a2;
  out[2 * EB + 1 + i] = mv;
  ss[tid] = mv;
  sq[tid] = mv * mv;
  __syncthreads();
  for (int st = 128; st > 0; st >>= 1) {
    if (tid < st) { ss[tid] += ss[tid + st]; sq[tid] += sq[tid + st]; }
    __syncthreads();
  }
  if (tid == 0) {
    partials[2 * blockIdx.x]     = ss[0];
    partials[2 * blockIdx.x + 1] = sq[0];
  }
}

__global__ __launch_bounds__(256)
void std_finalize(const float* __restrict__ partials, float* __restrict__ out,
                  int nparts) {
  __shared__ float ss[256], sq[256];
  int tid = threadIdx.x;
  float s = 0.0f, q = 0.0f;
  for (int i = tid; i < nparts; i += 256) {
    s += partials[2 * i];
    q += partials[2 * i + 1];
  }
  ss[tid] = s; sq[tid] = q;
  __syncthreads();
  for (int st = 128; st > 0; st >>= 1) {
    if (tid < st) { ss[tid] += ss[tid + st]; sq[tid] += sq[tid + st]; }
    __syncthreads();
  }
  if (tid == 0) {
    const float N = (float)EB;
    float mean = ss[0] / N;
    float var = (sq[0] - N * mean * mean) / (N - 1.0f);
    out[2 * EB] = sqrtf(fmaxf(var, 0.0f));
  }
}

// ---------------------------------------------------------------------------
extern "C" void kernel_launch(void* const* d_in, const int* in_sizes, int n_in,
                              void* d_out, int out_size, void* d_ws, size_t ws_size,
                              hipStream_t stream) {
  const float* state   = (const float*)d_in[0];
  const float* actions = (const float*)d_in[1];
  const float* W_in[2]  = {(const float*)d_in[2],  (const float*)d_in[8]};
  const float* b_in[2]  = {(const float*)d_in[3],  (const float*)d_in[9]};
  const float* W_hid[2] = {(const float*)d_in[4],  (const float*)d_in[10]};
  const float* b_hid[2] = {(const float*)d_in[5],  (const float*)d_in[11]};
  const float* W_out[2] = {(const float*)d_in[6],  (const float*)d_in[12]};
  const float* b_out[2] = {(const float*)d_in[7],  (const float*)d_in[13]};

  char* ws = (char*)d_ws;
  __bf16* xu  = (__bf16*)(ws + 0);                 // 26,214,400 B
  __bf16* h1  = (__bf16*)(ws + 26214400);          //  8,388,608 B
  __bf16* h2  = (__bf16*)(ws + 34603008);          //  8,388,608 B
  float*  res      = (float*)(ws + 42991616);      //    655,360 B
  float*  partials = (float*)(ws + 43646976);      //      4,096 B
  __bf16* W1t = (__bf16*)(ws + 43651072);          // 26,214,400 B  (40 x [H_ x D_])
  __bf16* W2t = (__bf16*)(ws + 69865472);          // 83,886,080 B  (40 x [H_ x H_])

  float* out = (float*)d_out;  // [a1(EB) | a2(EB) | std(1) | min(EB)]

  build_xu<<<2048, 256, 0, stream>>>(state, actions, xu);

  // Pre-convert + transpose all weights to bf16 [H x K] layout.
  for (int net = 0; net < 2; ++net) {
    convert_transpose<<<dim3(H_ / 64, D_ / 64, K_ * E_), 256, 0, stream>>>(
        W_in[net],  W1t + (size_t)net * K_ * E_ * H_ * D_, D_);
    convert_transpose<<<dim3(H_ / 64, H_ / 64, K_ * E_), 256, 0, stream>>>(
        W_hid[net], W2t + (size_t)net * K_ * E_ * H_ * H_, H_);
  }

  const dim3 ggrid(H_ / 128, B_ / 128);   // (8, 32)
  for (int net = 0; net < 2; ++net) {
    for (int k = 0; k < K_; ++k) {
      for (int e = 0; e < E_; ++e) {
        const size_t ke = (size_t)k * E_ + e;
        const size_t z  = (size_t)net * (K_ * E_) + ke;
        const __bf16* w1t = W1t + z * (size_t)H_ * D_;
        const __bf16* w2t = W2t + z * (size_t)H_ * H_;
        const float*  w3  = W_out[net] + ke * (size_t)H_;
        const float*  bb1 = b_in[net]  + ke * (size_t)H_;
        const float*  bb2 = b_hid[net] + ke * (size_t)H_;
        const float*  bb3 = b_out[net] + ke;

        gemm_wmma_async<<<ggrid, 256, 0, stream>>>(
            xu + (size_t)e * B_ * D_, D_, D_, w1t, bb1, h1);
        gemm_wmma_async<<<ggrid, 256, 0, stream>>>(
            h1, H_, H_, w2t, bb2, h2);
        head_out<<<B_ / 8, 256, 0, stream>>>(
            h2, w3, bb3, res + ((size_t)net * K_ + k) * EB + (size_t)e * B_);
      }
    }
  }

  combine<<<EB / 256, 256, 0, stream>>>(res, out, partials);
  std_finalize<<<1, 256, 0, stream>>>(partials, out, EB / 256);
}
